// MultiHeadAttention_15401752723638
// MI455X (gfx1250) — compile-verified, matrix operand order checked by reading, not run on hardware
//
#include <hip/hip_runtime.h>

// ---------------------------------------------------------------------------
// CDNA5 (gfx1250) single-head causal attention with MULTIPLICATIVE tril mask.
// B=4, T=2048, D=1024, fp32 in/out.  All GEMMs: V_WMMA_F32_16X16X32_BF16.
// Data movement: all fragment loads are contiguous 16-byte b128 ops (weights
// and V are pre-transposed); the scores GEMM stages Q/K tiles into LDS with
// the Tensor Data Mover (tensor_load_to_lds + s_wait_tensorcnt) when the
// toolchain exposes the builtin, else a cooperative b128 copy.
// ---------------------------------------------------------------------------

typedef __attribute__((ext_vector_type(16))) __bf16 v16bf;
typedef __attribute__((ext_vector_type(8)))  __bf16 bf16x8;
typedef __attribute__((ext_vector_type(8)))  float  v8f;
typedef __attribute__((ext_vector_type(4)))  float  f32x4;
typedef __attribute__((ext_vector_type(4)))  unsigned int u32x4;
typedef __attribute__((ext_vector_type(8)))  int    i32x8;
typedef __attribute__((ext_vector_type(4)))  int    i32x4;

union BF16x16 { v16bf v; __bf16 e[16]; bf16x8 h[2]; };
union F32x8   { v8f   v; float  e[8];  };

static constexpr int Bn = 4;
static constexpr int Tn = 2048;
static constexpr int Dn = 1024;

static __device__ __forceinline__ int lane_id() { return (int)(threadIdx.x & 31); }

// ---------------------------------------------------------------------------
// 16-bit 16x32 fragment load (A-layout; also the B-fragment of any matrix we
// hold transposed).  Lane L owns row (L&15); its 16 halves are two contiguous
// 8-element chunks at K = kb..kb+7 and kb+16..kb+23, kb = (L>=16)*8.
// Works for global or LDS source (compiler picks b128 / ds_load_b128).
// ---------------------------------------------------------------------------
static __device__ __forceinline__ v16bf load_frag_rows(const __bf16* tile, int ld) {
    int lane = lane_id();
    const __bf16* p = tile + (size_t)(lane & 15) * ld + ((lane >> 4) * 8);
    BF16x16 f;
    f.h[0] = *(const bf16x8*)p;         // K = kb .. kb+7
    f.h[1] = *(const bf16x8*)(p + 16);  // K = kb+16 .. kb+23
    return f.v;
}

// Same fragment from an fp32 source (converts inline; 4x b128 loads).
static __device__ __forceinline__ v16bf load_frag_rows_f32(const float* tile, int ld) {
    int lane = lane_id();
    const float* p = tile + (size_t)(lane & 15) * ld + ((lane >> 4) * 8);
    f32x4 u0 = *(const f32x4*)(p);
    f32x4 u1 = *(const f32x4*)(p + 4);
    f32x4 u2 = *(const f32x4*)(p + 16);
    f32x4 u3 = *(const f32x4*)(p + 20);
    BF16x16 f;
#pragma unroll
    for (int i = 0; i < 4; ++i) {
        f.e[i]      = (__bf16)u0[i];
        f.e[4 + i]  = (__bf16)u1[i];
        f.e[8 + i]  = (__bf16)u2[i];
        f.e[12 + i] = (__bf16)u3[i];
    }
    return f.v;
}

static __device__ __forceinline__ v8f wmma_bf16(v16bf a, v16bf b, v8f c) {
    return __builtin_amdgcn_wmma_f32_16x16x32_bf16(
        false, a, false, b, (short)0, c, false, false);
}

// ---------------------------------------------------------------------------
// Kernel 0: weight transpose + fp32->bf16 convert (tiled through LDS).
// Wt[n][k] = (bf16) W[k][n].  Grid: 1024 blocks (32x32 tiles), 256 threads.
// ---------------------------------------------------------------------------
__global__ void wt_kernel(const float* __restrict__ W, __bf16* __restrict__ Wt) {
    __shared__ float tile[32][33];
    int bx = blockIdx.x & 31;        // n-tile
    int by = blockIdx.x >> 5;        // k-tile
    int tx = threadIdx.x & 31;
    int ty = threadIdx.x >> 5;       // 0..7
#pragma unroll
    for (int i = 0; i < 4; ++i)
        tile[ty + 8 * i][tx] = W[(size_t)(by * 32 + ty + 8 * i) * Dn + bx * 32 + tx];
    __syncthreads();
#pragma unroll
    for (int i = 0; i < 4; ++i)
        Wt[(size_t)(bx * 32 + ty + 8 * i) * Dn + by * 32 + tx] =
            (__bf16)tile[tx][ty + 8 * i];
}

// ---------------------------------------------------------------------------
// Kernel 1: Q/K/V projections.  One wave per 16x16 tile; the x A-fragment is
// reused across all three (transposed) weights -> 3 WMMAs per k-step with
// purely contiguous b128 fragment loads.  Q,K stored row-major bf16; V is
// stored TRANSPOSED (Vt[b][d][t]) so the P@V B-fragments become contiguous.
// ---------------------------------------------------------------------------
__global__ void qkv_project_kernel(const float* __restrict__ x,
                                   const __bf16* __restrict__ Wtq,
                                   const __bf16* __restrict__ Wtk,
                                   const __bf16* __restrict__ Wtv,
                                   __bf16* __restrict__ Q,
                                   __bf16* __restrict__ K,
                                   __bf16* __restrict__ Vt) {
    int wave = blockIdx.x * (blockDim.x >> 5) + (threadIdx.x >> 5);
    int nt = wave & 63;              // D/16
    int mt = wave >> 6;              // B*T/16
    int nbase = nt * 16;
    const float* xa = x + (size_t)mt * 16 * Dn;

    v8f cq = {}; v8f ck = {}; v8f cv = {};
    for (int kk = 0; kk < Dn; kk += 32) {
        v16bf a  = load_frag_rows_f32(xa + kk, Dn);
        v16bf bq = load_frag_rows(Wtq + (size_t)nbase * Dn + kk, Dn);
        v16bf bk = load_frag_rows(Wtk + (size_t)nbase * Dn + kk, Dn);
        v16bf bv = load_frag_rows(Wtv + (size_t)nbase * Dn + kk, Dn);
        cq = wmma_bf16(a, bq, cq);
        ck = wmma_bf16(a, bk, ck);
        cv = wmma_bf16(a, bv, cv);
    }

    int lane  = lane_id();
    int col   = nbase + (lane & 15);
    int rbase = mt * 16 + ((lane >> 4) * 8);
    F32x8 q_; q_.v = cq;
    F32x8 k_; k_.v = ck;
    F32x8 v_; v_.v = cv;
#pragma unroll
    for (int r = 0; r < 8; ++r) {
        int row = rbase + r;                       // flat b*T + t
        size_t idx = (size_t)row * Dn + col;
        Q[idx] = (__bf16)q_.e[r];
        K[idx] = (__bf16)k_.e[r];
        int b = row >> 11, t = row & (Tn - 1);
        Vt[((size_t)b * Dn + col) * Tn + t] = (__bf16)v_.e[r];
    }
}

// ---------------------------------------------------------------------------
// TDM staging of a 2-D bf16 tile (tile_x cols x tile_y rows, row stride
// `ld` elements) from global into LDS.  Descriptor packed per ISA section 8:
// group0 = {count=1 | lds_addr | global_addr | type=2},
// group1 = {data_size=2B, tensor_dim0/1, tile_dim0/1, dim0 stride}.
// This toolchain's builtin takes SIX args: (g0, g1, g2, g3, g4, cpol).
// ---------------------------------------------------------------------------
#if defined(__HIP_DEVICE_COMPILE__) && __has_builtin(__builtin_amdgcn_tensor_load_to_lds) && \
    __has_builtin(__builtin_amdgcn_s_wait_tensorcnt)
#define CDNA5_USE_TDM 1
#endif

#if CDNA5_USE_TDM
static __device__ __forceinline__ void tdm_load_tile_2d(const __bf16* gsrc,
                                                        __bf16* lds_dst,
                                                        int tile_x, int tile_y, int ld) {
    unsigned long long ga = (unsigned long long)(uintptr_t)gsrc;
    u32x4 g0;
    g0[0] = 1u;                                        // count=1, user descriptor
    g0[1] = (unsigned)(uintptr_t)lds_dst;              // LDS byte address
    g0[2] = (unsigned)(ga & 0xFFFFFFFFu);              // global_addr[31:0]
    g0[3] = (unsigned)((ga >> 32) & 0x01FFFFFFu) | (2u << 30);  // addr[56:32] | type=2
    i32x8 g1;
    g1[0] = 1 << 16;                                   // data_size = 1 -> 2 bytes
    g1[1] = (int)(((unsigned)ld & 0xFFFFu) << 16);     // tensor_dim0[15:0]
    g1[2] = (int)((((unsigned)ld >> 16) & 0xFFFFu) |
                  (((unsigned)Tn & 0xFFFFu) << 16));   // tensor_dim0 hi | tensor_dim1 lo
    g1[3] = (int)(((unsigned)Tn >> 16) & 0xFFFFu) |
            (int)(((unsigned)tile_x & 0xFFFFu) << 16); // tensor_dim1 hi | tile_dim0
    g1[4] = tile_y & 0xFFFF;                           // tile_dim1 (tile_dim2 = 0)
    g1[5] = ld;                                        // tensor_dim0_stride[31:0]
    g1[6] = 0;
    g1[7] = 0;
    i32x4 z4 = {0, 0, 0, 0};
    i32x8 z8 = {0, 0, 0, 0, 0, 0, 0, 0};
    __builtin_amdgcn_tensor_load_to_lds(g0, g1, z4, z4, z8, 0);
}
#endif

// ---------------------------------------------------------------------------
// Kernel 2: masked scores, block-cooperative 64x64 macro-tile.
// 8 waves: wave w -> q-rows (w&3)*16, k-cols (w>>2)*32 (2 C-fragments).
// Per 32-deep d-step: stage Q(64x32) and K(64x32) bf16 tiles into LDS (TDM
// when available), then each wave issues 2 WMMAs from ds_load_b128 fragments.
// The K^T B-fragment is an A-style row load of K (free transpose).
// ---------------------------------------------------------------------------
__global__ void scores_kernel(const __bf16* __restrict__ Q,
                              const __bf16* __restrict__ K,
                              float* __restrict__ S) {
    const float gamma = 0.03125f;  // 1/sqrt(1024)
    __shared__ __align__(16) __bf16 sQ[64 * 32];
    __shared__ __align__(16) __bf16 sK[64 * 32];

    int blk = blockIdx.x;                 // b(4) x qt(32) x kt(32)
    int kt = blk & 31;
    int qt = (blk >> 5) & 31;
    int b  = blk >> 10;
    int qb  = qt * 64;
    int kb64 = kt * 64;
    float* Sb = S + (size_t)b * Tn * Tn;

    if (kt > qt) {  // whole 64x64 block strictly above diagonal -> zeros
        int r = threadIdx.x & 63;
        int c = (threadIdx.x >> 6) * 16;
        float* rowp = Sb + (size_t)(qb + r) * Tn + kb64 + c;
        f32x4 z = {0.f, 0.f, 0.f, 0.f};
#pragma unroll
        for (int i = 0; i < 16; i += 4) *(f32x4*)(rowp + i) = z;
        return;
    }

    const __bf16* Qp = Q + (size_t)(b * Tn + qb) * Dn;
    const __bf16* Kp = K + (size_t)(b * Tn + kb64) * Dn;

    int wid    = threadIdx.x >> 5;
    int wq_row = (wid & 3) * 16;
    int wk_col = (wid >> 2) * 32;

    v8f c0 = {}; v8f c1 = {};
    for (int d = 0; d < Dn; d += 32) {
        __syncthreads();   // previous iteration done reading sQ/sK
#if CDNA5_USE_TDM
        if (threadIdx.x < 32) {
            tdm_load_tile_2d(Qp + d, sQ, 32, 64, Dn);
            tdm_load_tile_2d(Kp + d, sK, 32, 64, Dn);
            __builtin_amdgcn_s_wait_tensorcnt(0);
        }
#else
        {
            int t = threadIdx.x;          // 256 threads, 16B chunks
            int row = t >> 2, off = (t & 3) * 8;
            *(bf16x8*)(sQ + row * 32 + off) = *(const bf16x8*)(Qp + d + (size_t)row * Dn + off);
            *(bf16x8*)(sK + row * 32 + off) = *(const bf16x8*)(Kp + d + (size_t)row * Dn + off);
        }
#endif
        __syncthreads();

        v16bf a  = load_frag_rows(sQ + wq_row * 32, 32);
        v16bf b0 = load_frag_rows(sK + wk_col * 32, 32);         // B-frag of K^T
        v16bf b1 = load_frag_rows(sK + (wk_col + 16) * 32, 32);
        c0 = wmma_bf16(a, b0, c0);
        c1 = wmma_bf16(a, b1, c1);
    }

    int lane  = lane_id();
    int rbase = qb + wq_row + ((lane >> 4) * 8);
    int j0    = kb64 + wk_col + (lane & 15);
    F32x8 f0; f0.v = c0;
    F32x8 f1; f1.v = c1;
#pragma unroll
    for (int r = 0; r < 8; ++r) {
        int i = rbase + r;
        float s0 = f0.e[r] * gamma; if (j0 > i)      s0 = 0.0f;  // multiplicative mask
        float s1 = f1.e[r] * gamma; if (j0 + 16 > i) s1 = 0.0f;
        Sb[(size_t)i * Tn + j0]      = s0;
        Sb[(size_t)i * Tn + j0 + 16] = s1;
    }
}

// ---------------------------------------------------------------------------
// Kernel 3: plain row softmax (zeros from the multiplicative mask participate,
// exactly matching the reference).  One 256-thread block per row, bf16 out.
// ---------------------------------------------------------------------------
__global__ void softmax_rows_kernel(const float* __restrict__ S,
                                    __bf16* __restrict__ P) {
    __shared__ float red[256];
    size_t row = blockIdx.x;
    const float* s = S + row * Tn;
    __bf16* p = P + row * Tn;
    int tid = threadIdx.x;

    float vals[8];
    float m = -__builtin_inff();
#pragma unroll
    for (int t = 0; t < 8; ++t) {
        vals[t] = s[tid + t * 256];
        m = fmaxf(m, vals[t]);
    }
    red[tid] = m;
    __syncthreads();
    for (int off = 128; off > 0; off >>= 1) {
        if (tid < off) red[tid] = fmaxf(red[tid], red[tid + off]);
        __syncthreads();
    }
    m = red[0];
    __syncthreads();

    float sum = 0.0f;
#pragma unroll
    for (int t = 0; t < 8; ++t) {
        vals[t] = __expf(vals[t] - m);
        sum += vals[t];
    }
    red[tid] = sum;
    __syncthreads();
    for (int off = 128; off > 0; off >>= 1) {
        if (tid < off) red[tid] += red[tid + off];
        __syncthreads();
    }
    float inv = 1.0f / red[0];
#pragma unroll
    for (int t = 0; t < 8; ++t)
        p[tid + t * 256] = (__bf16)(vals[t] * inv);
}

// ---------------------------------------------------------------------------
// Kernel 4: Y = P @ V via Vt.  One wave per 32x32 output macro-tile: 4 WMMAs
// per 32-deep j-step against 8 contiguous b128 fragment loads (A from P rows,
// B from Vt rows -- no gathers anywhere).
// ---------------------------------------------------------------------------
__global__ void out_kernel(const __bf16* __restrict__ P,
                           const __bf16* __restrict__ Vt,
                           float* __restrict__ Y) {
    int wave = blockIdx.x * (blockDim.x >> 5) + (threadIdx.x >> 5);
    int nt = wave & 31;               // D/32
    int qt = (wave >> 5) & 63;        // T/32
    int b  = wave >> 11;
    int qbase = qt * 32, nbase = nt * 32;

    const __bf16* Pp  = P  + (size_t)(b * Tn + qbase) * Tn;
    const __bf16* Vp  = Vt + ((size_t)b * Dn + nbase) * Tn;

    v8f c00 = {}; v8f c01 = {}; v8f c10 = {}; v8f c11 = {};
    for (int j = 0; j < Tn; j += 32) {
        v16bf a0 = load_frag_rows(Pp + j, Tn);
        v16bf a1 = load_frag_rows(Pp + (size_t)16 * Tn + j, Tn);
        v16bf b0 = load_frag_rows(Vp + j, Tn);
        v16bf b1 = load_frag_rows(Vp + (size_t)16 * Tn + j, Tn);
        c00 = wmma_bf16(a0, b0, c00);
        c01 = wmma_bf16(a0, b1, c01);
        c10 = wmma_bf16(a1, b0, c10);
        c11 = wmma_bf16(a1, b1, c11);
    }

    int lane  = lane_id();
    int col   = nbase + (lane & 15);
    int rbase = qbase + ((lane >> 4) * 8);
    F32x8 f00; f00.v = c00;
    F32x8 f01; f01.v = c01;
    F32x8 f10; f10.v = c10;
    F32x8 f11; f11.v = c11;
#pragma unroll
    for (int r = 0; r < 8; ++r) {
        size_t row0 = (size_t)(b * Tn + rbase + r) * Dn;
        size_t row1 = (size_t)(b * Tn + rbase + 16 + r) * Dn;
        Y[row0 + col]      = f00.e[r];
        Y[row0 + col + 16] = f01.e[r];
        Y[row1 + col]      = f10.e[r];
        Y[row1 + col + 16] = f11.e[r];
    }
}

// ---------------------------------------------------------------------------
// Workspace layout (128 MB):
//   [  0,  16M) Q bf16          (later overwritten by P, first half)
//   [ 16,  32M) K bf16          (later overwritten by P, second half)
//   [ 32,  48M) Vt bf16  [B][D][T]
//   [ 48,  50M) Wtq bf16 [D][D]
//   [ 50,  52M) Wtk bf16
//   [ 52,  54M) Wtv bf16
//   [ 64, 128M) S fp32   [B][T][T]
//   P bf16 (32M) reuses [0, 32M) after scores_kernel completes.
// ---------------------------------------------------------------------------
extern "C" void kernel_launch(void* const* d_in, const int* in_sizes, int n_in,
                              void* d_out, int out_size, void* d_ws, size_t ws_size,
                              hipStream_t stream) {
    const float* x  = (const float*)d_in[0];
    const float* wq = (const float*)d_in[1];
    const float* wk = (const float*)d_in[2];
    const float* wv = (const float*)d_in[3];
    float* out = (float*)d_out;

    char* ws = (char*)d_ws;
    const size_t MB = 1024ull * 1024ull;
    __bf16* Qb  = (__bf16*)(ws + 0 * MB);
    __bf16* Kb  = (__bf16*)(ws + 16 * MB);
    __bf16* Vt  = (__bf16*)(ws + 32 * MB);
    __bf16* Wtq = (__bf16*)(ws + 48 * MB);
    __bf16* Wtk = (__bf16*)(ws + 50 * MB);
    __bf16* Wtv = (__bf16*)(ws + 52 * MB);
    float*  S   = (float*)(ws + 64 * MB);
    __bf16* Pb  = (__bf16*)(ws + 0 * MB);   // reuse Q/K region

    // 0) transpose+convert the three weight matrices (32x32 LDS tiles)
    wt_kernel<<<1024, 256, 0, stream>>>(wq, Wtq);
    wt_kernel<<<1024, 256, 0, stream>>>(wk, Wtk);
    wt_kernel<<<1024, 256, 0, stream>>>(wv, Wtv);
    // 1) projections: 512x64 tiles / 8 waves per block
    qkv_project_kernel<<<4096, 256, 0, stream>>>(x, Wtq, Wtk, Wtv, Qb, Kb, Vt);
    // 2) masked scores: 64x64 block tiles, TDM-staged LDS
    scores_kernel<<<4096, 256, 0, stream>>>(Qb, Kb, S);
    // 3) row softmax: one block per row
    softmax_rows_kernel<<<Bn * Tn, 256, 0, stream>>>(S, Pb);
    // 4) Y = P @ V: 32x32 per-wave macro tiles
    out_kernel<<<1024, 256, 0, stream>>>(Pb, Vt, out);
}
